// MultiHeadAttention_1898375544924
// MI455X (gfx1250) — matrix operand order repaired and checked by reading, not compiled, not run on hardware
//
#include <hip/hip_runtime.h>
#include <hip/hip_bf16.h>

// MI455X / gfx1250 multi-head attention forward.
// B=2, S=2048, D=1024, H=16, HD=64.
// All GEMMs on v_wmma_f32_16x16x32_f16; V tiles staged with
// global_load_async_to_lds_b128 (+ s_wait_asynccnt).

typedef __attribute__((ext_vector_type(16))) _Float16 v16h;
typedef __attribute__((ext_vector_type(8)))  float    v8f;

#define BATCH  2
#define SEQ    2048
#define DMODEL 1024
#define HEADS  16
#define HDIM   64
#define BH     (BATCH * HEADS)
#define MROWS  (BATCH * SEQ)   // 4096

__device__ __forceinline__ v8f wmma_f16(v16h a, v16h b, v8f c) {
  // (neg_a, A, neg_b, B, c_mod, C, reuse_a, reuse_b)
  return __builtin_amdgcn_wmma_f32_16x16x32_f16(false, a, false, b, (short)0, c,
                                                false, false);
}

// ---- Fragment builders (contiguous k order: K = 16*(lane/16) + e, same for A and B) ----

// A fragment 16x32 from row-major LDS tile (leading dim ld halfs).
// Reads are one 32-byte contiguous run per lane -> ds_load_b128 x2.
__device__ __forceinline__ v16h frag_a_lds(const _Float16* base, int ld, int lane) {
  const int row = lane & 15;
  const int kb  = (lane & 16) ? 16 : 0;  // lanes 16-31: K=16..31
  const _Float16* p = base + row * ld + kb;
  v16h a;
#pragma unroll
  for (int e = 0; e < 16; ++e) a[e] = p[e];
  return a;
}

// B fragment 32x16 from a TRANSPOSED LDS tile Bt[col][k] (leading dim ld).
// Element e <-> K = kb+e, contiguous in k -> ds_load_b128 x2 per lane.
__device__ __forceinline__ v16h frag_b_ldsT(const _Float16* base, int ld, int lane) {
  const int col = lane & 15;
  const int kb  = (lane & 16) ? 16 : 0;  // lanes 16-31: K=16..31
  const _Float16* p = base + col * ld + kb;
  v16h b;
#pragma unroll
  for (int e = 0; e < 16; ++e) b[e] = p[e];
  return b;
}

// B fragment 32x16 from a natural-layout LDS tile Bs[k][col] (leading dim ld).
__device__ __forceinline__ v16h frag_b_lds(const _Float16* base, int ld, int lane) {
  const int col = lane & 15;
  const int kb  = (lane & 16) ? 16 : 0;
  const _Float16* p = base + kb * ld + col;
  v16h b;
#pragma unroll
  for (int e = 0; e < 16; ++e) b[e] = p[e * ld];
  return b;
}

// A fragment straight from global row-major [*,64] (Q), k-offset ks.
__device__ __forceinline__ v16h frag_a_glb(const _Float16* base, int ks, int lane) {
  const int row = lane & 15;
  const int kb  = ks + ((lane & 16) ? 16 : 0);
  const _Float16* p = base + row * HDIM + kb;
  v16h a;
#pragma unroll
  for (int e = 0; e < 16; ++e) a[e] = p[e];
  return a;
}

// B = K^T fragment straight from global: B[k][n] = Kmat[n*64+k], contiguous in k.
__device__ __forceinline__ v16h frag_bT_glb(const _Float16* base, int ks, int lane) {
  const int col = lane & 15;
  const int kb  = ks + ((lane & 16) ? 16 : 0);
  const _Float16* p = base + col * HDIM + kb;
  v16h b;
#pragma unroll
  for (int e = 0; e < 16; ++e) b[e] = p[e];
  return b;
}

// CDNA5 async LDS-direct 16-byte load (ASYNCcnt-tracked).
__device__ __forceinline__ void async_ld_b128(unsigned lds_addr,
                                              unsigned long long gaddr) {
  asm volatile("global_load_async_to_lds_b128 %0, %1, off"
               :: "v"(lds_addr), "v"(gaddr) : "memory");
}
__device__ __forceinline__ void wait_asynccnt0() {
  asm volatile("s_wait_asynccnt 0x0" ::: "memory");
}

// ---------------- Kernel 1: QKV = X @ W_qkv, scatter to head-major f16 -----
// Macro-tile 32x128; 8 waves as 2x4; each wave: 1 A-frag x 2 B-frags.

__global__ __launch_bounds__(256) void qkv_gemm_kernel(
    const float* __restrict__ X, const float* __restrict__ W,
    _Float16* __restrict__ Qd, _Float16* __restrict__ Kd, _Float16* __restrict__ Vd) {
  __shared__ __align__(16) _Float16 As[32][40];    // row-major [m][k]
  __shared__ __align__(16) _Float16 Bt[128][40];   // transposed [n][k]
  const int tid = threadIdx.x, lane = tid & 31, wave = tid >> 5;
  const int wr = wave >> 2, wc = wave & 3;
  const int row0 = blockIdx.y * 32;
  const int col0 = blockIdx.x * 128;
  v8f acc[2] = {{}, {}};
  for (int kt = 0; kt < DMODEL; kt += 32) {
#pragma unroll
    for (int i = 0; i < 4; ++i) {                  // 32x32 A, f32->f16
      int idx = i * 256 + tid, r = idx >> 5, c = idx & 31;
      As[r][c] = (_Float16)X[(size_t)(row0 + r) * DMODEL + kt + c];
    }
#pragma unroll
    for (int i = 0; i < 16; ++i) {                 // 32x128 B, f32->f16, transposed
      int idx = i * 256 + tid, r = idx >> 7, c = idx & 127;
      Bt[c][r] = (_Float16)W[(size_t)(kt + r) * (3 * DMODEL) + col0 + c];
    }
    __syncthreads();
    v16h a = frag_a_lds(&As[wr * 16][0], 40, lane);
#pragma unroll
    for (int f = 0; f < 2; ++f) {
      v16h b = frag_b_ldsT(&Bt[wc * 32 + f * 16][0], 40, lane);
      acc[f] = wmma_f16(a, b, acc[f]);
    }
    __syncthreads();
  }
  const int mb = (lane & 16) ? 8 : 0;
#pragma unroll
  for (int f = 0; f < 2; ++f) {
    const int coln = col0 + wc * 32 + f * 16 + (lane & 15);  // 0..3071
    const int part = coln >> 10;                              // 0=Q 1=K 2=V
    const int dcol = coln & (DMODEL - 1);
    const int h = dcol >> 6, hd = dcol & 63;
    _Float16* dst = (part == 0) ? Qd : ((part == 1) ? Kd : Vd);
#pragma unroll
    for (int v = 0; v < 8; ++v) {
      int m = row0 + wr * 16 + mb + v;
      int b = m >> 11, s = m & (SEQ - 1);
      dst[(((size_t)(b * HEADS + h)) * SEQ + s) * HDIM + hd] = (_Float16)acc[f][v];
    }
  }
}

// ---------------- Kernel 2: raw causal scores = Q K^T / sqrt(64) -----------

__global__ __launch_bounds__(256) void scores_kernel(
    const _Float16* __restrict__ Q, const _Float16* __restrict__ Kh,
    float* __restrict__ attn) {
  const int lane = threadIdx.x & 31, wave = threadIdx.x >> 5;
  const int bh = blockIdx.z;
  const int rt = blockIdx.y;                 // row tile 0..127
  const int ct = blockIdx.x * 8 + wave;      // col tile, one per wave
  if (ct > rt) return;                       // strictly-upper tiles are dead
  const _Float16* qb = Q  + ((size_t)bh * SEQ + rt * 16) * HDIM;
  const _Float16* kb = Kh + ((size_t)bh * SEQ + ct * 16) * HDIM;
  v8f acc = {};
#pragma unroll
  for (int ks = 0; ks < HDIM; ks += 32) {
    v16h a = frag_a_glb(qb, ks, lane);
    v16h b = frag_bT_glb(kb, ks, lane);
    acc = wmma_f16(a, b, acc);
  }
  const int mb = (lane & 16) ? 8 : 0;
  const int n  = ct * 16 + (lane & 15);
  float* orow = attn + (size_t)bh * SEQ * SEQ;
#pragma unroll
  for (int v = 0; v < 8; ++v) {
    int r = rt * 16 + mb + v;
    float val = acc[v] * 0.125f;             // 1/sqrt(64)
    if (n > r) val = -3.0e38f;               // causal mask (diagonal tiles)
    orow[(size_t)r * SEQ + n] = val;
  }
}

// ---------------- Kernel 3: row softmax in place (zeros above diagonal) ----

__global__ __launch_bounds__(256) void softmax_kernel(float* __restrict__ attn) {
  __shared__ float red[256];
  const int tid = threadIdx.x;
  const int r   = blockIdx.x;
  const int bh  = blockIdx.y;
  float* row = attn + ((size_t)bh * SEQ + r) * SEQ;
  const int nv = r + 1;
  float m = -3.0e38f;
  for (int c = tid; c < nv; c += 256) m = fmaxf(m, row[c]);
  red[tid] = m; __syncthreads();
#pragma unroll
  for (int s = 128; s > 0; s >>= 1) {
    if (tid < s) red[tid] = fmaxf(red[tid], red[tid + s]);
    __syncthreads();
  }
  const float mx = red[0]; __syncthreads();
  float sum = 0.0f;
  for (int c = tid; c < nv; c += 256) sum += __expf(row[c] - mx);
  red[tid] = sum; __syncthreads();
#pragma unroll
  for (int s = 128; s > 0; s >>= 1) {
    if (tid < s) red[tid] += red[tid + s];
    __syncthreads();
  }
  const float inv = 1.0f / red[0];
  __syncthreads();
  for (int c = tid; c < SEQ; c += 256)
    row[c] = (c < nv) ? __expf(row[c] - mx) * inv : 0.0f;
}

// ---------------- Kernel 4: context = P @ V ---------------------------------
// Macro-tile 64x64; 8 waves as 4x2; wave: 1 A-frag x 2 B-frags.
// V tile staged via global_load_async_to_lds_b128; K-loop stops at diagonal.

__global__ __launch_bounds__(256) void pv_gemm_kernel(
    const float* __restrict__ P, const _Float16* __restrict__ V,
    _Float16* __restrict__ ctx) {
  __shared__ __align__(16) _Float16 As[64][40];   // P tile  [m][k]
  __shared__ __align__(16) _Float16 Vs[32][72];   // V tile  [k][n] (natural)
  const int tid = threadIdx.x, lane = tid & 31, wave = tid >> 5;
  const int wr = wave >> 1, wc = wave & 1;        // 4x2 waves
  const int bh = blockIdx.y;
  const int rm = blockIdx.x * 64;
  const float*    Pb = P + (size_t)bh * SEQ * SEQ;
  const _Float16* Vb = V + (size_t)bh * SEQ * HDIM;
  v8f acc[2] = {{}, {}};
  const int kend = rm + 64;                       // P zero beyond the diagonal
  const int vr = tid >> 3, vchunk = tid & 7;      // 32 rows x 8 x 16B chunks
  const unsigned vlds = (unsigned)(size_t)&Vs[vr][vchunk * 8];
  for (int kt = 0; kt < kend; kt += 32) {
    // async LDS-direct copy of the V tile (one b128 per thread)
    async_ld_b128(vlds,
        (unsigned long long)(size_t)(Vb + (size_t)(kt + vr) * HDIM + vchunk * 8));
#pragma unroll
    for (int i = 0; i < 8; ++i) {                 // 64x32 P tile, f32->f16
      int idx = i * 256 + tid, r = idx >> 5, c = idx & 31;
      As[r][c] = (_Float16)Pb[(size_t)(rm + r) * SEQ + kt + c];
    }
    wait_asynccnt0();
    __syncthreads();
    v16h a = frag_a_lds(&As[wr * 16][0], 40, lane);
#pragma unroll
    for (int f = 0; f < 2; ++f) {
      v16h b = frag_b_lds(&Vs[0][wc * 32 + f * 16], 72, lane);
      acc[f] = wmma_f16(a, b, acc[f]);
    }
    __syncthreads();
  }
  const int mb = (lane & 16) ? 8 : 0;
  const int b = bh >> 4, h = bh & 15;
#pragma unroll
  for (int f = 0; f < 2; ++f) {
    const int n = wc * 32 + f * 16 + (lane & 15);  // 0..63 (head dim)
#pragma unroll
    for (int v = 0; v < 8; ++v) {
      int s = rm + wr * 16 + mb + v;
      ctx[((size_t)(b * SEQ + s)) * DMODEL + h * HDIM + n] = (_Float16)acc[f][v];
    }
  }
}

// ---------------- Kernel 5: out = context @ W_out + b_out (f32) ------------
// Macro-tile 32x128; 8 waves as 2x4; each wave: 1 A-frag x 2 B-frags.

__global__ __launch_bounds__(256) void out_gemm_kernel(
    const _Float16* __restrict__ Cx, const float* __restrict__ W,
    const float* __restrict__ bias, float* __restrict__ out) {
  __shared__ __align__(16) _Float16 As[32][40];
  __shared__ __align__(16) _Float16 Bt[128][40];  // transposed [n][k]
  const int tid = threadIdx.x, lane = tid & 31, wave = tid >> 5;
  const int wr = wave >> 2, wc = wave & 3;
  const int row0 = blockIdx.y * 32, col0 = blockIdx.x * 128;
  v8f acc[2] = {{}, {}};
  for (int kt = 0; kt < DMODEL; kt += 32) {
#pragma unroll
    for (int i = 0; i < 4; ++i) {                 // context tile (f16 copy)
      int idx = i * 256 + tid, r = idx >> 5, c = idx & 31;
      As[r][c] = Cx[(size_t)(row0 + r) * DMODEL + kt + c];
    }
#pragma unroll
    for (int i = 0; i < 16; ++i) {                // W_out tile f32->f16, transposed
      int idx = i * 256 + tid, r = idx >> 7, c = idx & 127;
      Bt[c][r] = (_Float16)W[(size_t)(kt + r) * DMODEL + col0 + c];
    }
    __syncthreads();
    v16h a = frag_a_lds(&As[wr * 16][0], 40, lane);
#pragma unroll
    for (int f = 0; f < 2; ++f) {
      v16h b = frag_b_ldsT(&Bt[wc * 32 + f * 16][0], 40, lane);
      acc[f] = wmma_f16(a, b, acc[f]);
    }
    __syncthreads();
  }
  const int mb = (lane & 16) ? 8 : 0;
#pragma unroll
  for (int f = 0; f < 2; ++f) {
    const int n = col0 + wc * 32 + f * 16 + (lane & 15);
    const float bn = bias[n];
#pragma unroll
    for (int v = 0; v < 8; ++v) {
      int m = row0 + wr * 16 + mb + v;
      out[(size_t)m * DMODEL + n] = acc[f][v] + bn;
    }
  }
}

// ---------------------------------------------------------------------------

extern "C" void kernel_launch(void* const* d_in, const int* in_sizes, int n_in,
                              void* d_out, int out_size, void* d_ws, size_t ws_size,
                              hipStream_t stream) {
  (void)in_sizes; (void)n_in; (void)out_size; (void)ws_size;
  const float* x    = (const float*)d_in[0];
  // d_in[1] = mask (causal, hardcoded in kernels)
  const float* Wqkv = (const float*)d_in[2];
  const float* Wout = (const float*)d_in[3];
  const float* bout = (const float*)d_in[4];

  float* outC = (float*)d_out;                                  // [B,S,D]
  float* attn = outC + (size_t)BATCH * SEQ * DMODEL;            // [B,H,S,S]

  _Float16* ws = (_Float16*)d_ws;
  const size_t qkvElems = (size_t)BATCH * HEADS * SEQ * HDIM;   // 4,194,304
  _Float16* Qh = ws;
  _Float16* Kh = Qh + qkvElems;
  _Float16* Vh = Kh + qkvElems;
  _Float16* Cx = Vh + qkvElems;                                 // [B,S,D] f16

  qkv_gemm_kernel<<<dim3(3 * DMODEL / 128, MROWS / 32), 256, 0, stream>>>(
      x, Wqkv, Qh, Kh, Vh);
  scores_kernel<<<dim3(SEQ / 16 / 8, SEQ / 16, BH), 256, 0, stream>>>(
      Qh, Kh, attn);
  softmax_kernel<<<dim3(SEQ, BH), 256, 0, stream>>>(attn);
  pv_gemm_kernel<<<dim3(SEQ / 64, BH), 256, 0, stream>>>(attn, Vh, Cx);
  out_gemm_kernel<<<dim3(DMODEL / 128, MROWS / 32), 256, 0, stream>>>(
      Cx, Wout, bout, outC);
}
